// RecursiveDecoder_8718783611512
// MI455X (gfx1250) — matrix operand order repaired and checked by reading, not compiled, not run on hardware
//
#include <hip/hip_runtime.h>

typedef __attribute__((ext_vector_type(16))) _Float16 v16h;
typedef __attribute__((ext_vector_type(8)))  _Float16 v8h;
typedef __attribute__((ext_vector_type(8)))  float    v8f;

constexpr int NCH  = 128;   // children
constexpr int HID  = 512;   // hidden
constexpr int FEA  = 512;   // node feat
constexpr int TE   = 4;     // edge types
constexpr int NSEM = 50;
constexpr int LDOP = 3*HID + TE; // 1540

// ---- WMMA operand loaders (CDNA5 16x16x32 f16 layouts) ----
// A-matrix (MxK=16x32): lane m=l&15, half=l>>4; K = e + (e>=8?8:0) + 8*half
__device__ __forceinline__ v16h load_tileA(const _Float16* p /* &row[m] + k0 + 8*half */) {
  v8h lo = *(const v8h*)(p);
  v8h hi = *(const v8h*)(p + 16);
  v16h a;
#pragma unroll
  for (int e = 0; e < 8; ++e) { a[e] = lo[e]; a[e + 8] = hi[e]; }
  return a;
}
// B-matrix (KxN=32x16) fed from a row-major data row (we compute data@W.T):
// lane n=l&15, half=l>>4; K = e + (e>=8?8:0) + 8*half
__device__ __forceinline__ v16h load_tileB(const _Float16* p /* &row[n] + k0 + 8*half */) {
  v8h lo = *(const v8h*)(p);
  v8h hi = *(const v8h*)(p + 16);
  v16h b;
#pragma unroll
  for (int e = 0; e < 8; ++e) { b[e] = lo[e]; b[e + 8] = hi[e]; }
  return b;
}

// ---- generic WMMA GEMM: C[row][hout] = A[row,:K] dot Bw[hout,:K] (+bias)(+relu)
// grid.x = Nfeat/128 (8 waves x 16 hout), grid.y = rows/16, block = 256
__global__ void gemm_wmma_kernel(const _Float16* __restrict__ A, int lda,
                                 const _Float16* __restrict__ Bw, int ldb,
                                 const float* __restrict__ bias,
                                 float* __restrict__ Cf, _Float16* __restrict__ Ch,
                                 int Nfeat, int K, int relu) {
  int wave = threadIdx.x >> 5, lane = threadIdx.x & 31;
  int ln = lane & 15, half = lane >> 4;
  int m0 = blockIdx.x * 128 + wave * 16;  // out-feature tile (WMMA M)
  int n0 = blockIdx.y * 16;               // data-row tile (WMMA N)
  v8f d;
#pragma unroll
  for (int r = 0; r < 8; ++r) d[r] = 0.0f;
  const _Float16* pw = Bw + (size_t)(m0 + ln) * ldb + 8 * half;
  const _Float16* pd = A  + (size_t)(n0 + ln) * lda + 8 * half;
  __builtin_prefetch(pw, 0, 3);
  __builtin_prefetch(pd, 0, 3);
  for (int k0 = 0; k0 < K; k0 += 32) {
    v16h a = load_tileA(pw + k0);
    v16h b = load_tileB(pd + k0);
    d = __builtin_amdgcn_wmma_f32_16x16x32_f16(false, a, false, b, (short)0, d, false, false);
  }
#pragma unroll
  for (int r = 0; r < 8; ++r) {
    int hout = m0 + r + 8 * half;
    int row  = n0 + ln;
    float v = d[r];
    if (bias) v += bias[hout];
    if (relu) v = fmaxf(v, 0.0f);
    size_t o = (size_t)row * Nfeat + hout;
    if (Cf) Cf[o] = v;
    if (Ch) Ch[o] = (_Float16)v;
  }
}

// ---- cf = relu(pf @ Wp.T + bp): one wave per output element ----
__global__ void cf_kernel(const float* __restrict__ pf, const float* __restrict__ Wp,
                          const float* __restrict__ bp,
                          float* __restrict__ cfF, _Float16* __restrict__ cfH) {
  int waveId = (int)((blockIdx.x * blockDim.x + threadIdx.x) >> 5);
  int lane = threadIdx.x & 31;
  const float* w = Wp + (size_t)waveId * FEA;
  float s = 0.f;
  for (int k = lane; k < FEA; k += 32) s += pf[k] * w[k];
#pragma unroll
  for (int off = 16; off; off >>= 1) s += __shfl_down(s, off, 32);
  if (lane == 0) {
    float v = fmaxf(s + bp[waveId], 0.f);
    cfF[waveId] = v; cfH[waveId] = (_Float16)v;
  }
}

__global__ void exists_kernel(const float* __restrict__ cfF, const float* __restrict__ We,
                              const float* __restrict__ be,
                              float* __restrict__ outEx, int* __restrict__ exflag) {
  int i = threadIdx.x;  // blockDim = 128
  float s = 0.f;
  for (int h = 0; h < HID; ++h) s += cfF[i * HID + h] * We[h];
  s += be[0];
  outEx[i] = s;
  exflag[i] = (s > 0.f) ? 1 : 0;
}

// el16[i,j,h] = relu(Af[i,h] + Bf[j,h])   (bel folded into Af's GEMM bias)
__global__ void el_kernel(const float* __restrict__ Af, const float* __restrict__ Bf,
                          _Float16* __restrict__ el16) {
  size_t idx = (size_t)blockIdx.x * 256 + threadIdx.x;  // < 128*128*512
  int h = (int)(idx & (HID - 1));
  int j = (int)((idx >> 9) & (NCH - 1));
  int i = (int)(idx >> 16);
  float v = Af[i * HID + h] + Bf[j * HID + h];
  el16[idx] = (_Float16)fmaxf(v, 0.f);
}

// eel[i,j,t] = el[i,j,:] . Wee[t,:] + bee[t]; also has_edges
__global__ void eel_kernel(const _Float16* __restrict__ el16, const float* __restrict__ Wee,
                           const float* __restrict__ bee, const int* __restrict__ exflag,
                           float* __restrict__ outEel, int* __restrict__ hasEdges) {
  int pair = (int)((blockIdx.x * blockDim.x + threadIdx.x) >> 5);  // < 16384
  int lane = threadIdx.x & 31;
  const _Float16* row = el16 + (size_t)pair * HID;
  float a0 = 0, a1 = 0, a2 = 0, a3 = 0;
  for (int k = lane; k < HID; k += 32) {
    float v = (float)row[k];
    a0 += v * Wee[k]; a1 += v * Wee[HID + k];
    a2 += v * Wee[2 * HID + k]; a3 += v * Wee[3 * HID + k];
  }
#pragma unroll
  for (int off = 16; off; off >>= 1) {
    a0 += __shfl_down(a0, off, 32); a1 += __shfl_down(a1, off, 32);
    a2 += __shfl_down(a2, off, 32); a3 += __shfl_down(a3, off, 32);
  }
  if (lane == 0) {
    int i = pair >> 7, j = pair & 127;
    float e[4] = {a0 + bee[0], a1 + bee[1], a2 + bee[2], a3 + bee[3]};
    int exij = exflag[i] && exflag[j];
    int any = 0;
#pragma unroll
    for (int t = 0; t < TE; ++t) { outEel[(size_t)pair * TE + t] = e[t]; any |= (exij && e[t] > 0.f); }
    if (any) atomicOr(hasEdges, 1);
  }
}

// ---- fused message-passing iteration: batched WMMA GEMM (el@Wf3.T) + masked max ----
// grid = (128 i, 4 h-chunks), block 256 (8 waves x 16 h). Wave M-dim = h, N-dim = j.
// k-outer loop with 8 live j-tile accumulators: A (Wf3) loaded once per k-step,
// ~1.1 operand loads per WMMA instead of 4.
__global__ void msg_max_kernel(const _Float16* __restrict__ el16,
                               const _Float16* __restrict__ Wf3h,
                               const float* __restrict__ P,     // includes bop bias
                               const float* __restrict__ Q,
                               const float* __restrict__ WopIt, // row stride LDOP, Wt at col 3H+t
                               const float* __restrict__ eel,
                               const int* __restrict__ exflag,
                               const int* __restrict__ hasEdges,
                               const float* __restrict__ c0prev,
                               float* __restrict__ outF, _Float16* __restrict__ outH) {
  int i = blockIdx.x;
  int wave = threadIdx.x >> 5, lane = threadIdx.x & 31;
  int ln = lane & 15, half = lane >> 4;
  int h0 = blockIdx.y * 128 + wave * 16;

  const _Float16* pa = Wf3h + (size_t)(h0 + ln) * HID + 8 * half;
  const _Float16* pb = el16 + ((size_t)i * NCH + ln) * HID + 8 * half;
  __builtin_prefetch(pa, 0, 3);
  __builtin_prefetch(pb, 0, 3);

  v8f dacc[8];
#pragma unroll
  for (int jt = 0; jt < 8; ++jt)
#pragma unroll
    for (int r = 0; r < 8; ++r) dacc[jt][r] = 0.0f;

  for (int k0 = 0; k0 < HID; k0 += 32) {
    v16h a = load_tileA(pa + k0);
#pragma unroll
    for (int jt = 0; jt < 8; ++jt) {
      v16h b = load_tileB(pb + k0 + (size_t)jt * (16 * HID));
      dacc[jt] = __builtin_amdgcn_wmma_f32_16x16x32_f16(false, a, false, b, (short)0,
                                                        dacc[jt], false, false);
    }
  }

  // ---- epilogue: masked relu+max over (j, t) ----
  float Pv[8], Wt[8][4];
#pragma unroll
  for (int r = 0; r < 8; ++r) {
    int h = h0 + r + 8 * half;
    Pv[r] = P[i * HID + h];
#pragma unroll
    for (int t = 0; t < TE; ++t) Wt[r][t] = WopIt[(size_t)h * LDOP + 3 * HID + t];
  }
  int exi = exflag[i];
  float acc[8];
#pragma unroll
  for (int r = 0; r < 8; ++r) acc[r] = 0.0f;

#pragma unroll
  for (int jt = 0; jt < 8; ++jt) {
    int j = jt * 16 + ln;  // this lane's column (WMMA N)
    float e0t[4]; int mt[4];
    int exj = exflag[j];
#pragma unroll
    for (int t = 0; t < TE; ++t) {
      e0t[t] = eel[(size_t)(i * NCH + j) * TE + t];
      mt[t] = (exi && exj && (e0t[t] > 0.f)) ? 1 : 0;
    }
#pragma unroll
    for (int r = 0; r < 8; ++r) {
      int h = h0 + r + 8 * half;
      float base = dacc[jt][r] + Pv[r] + Q[(size_t)j * HID + h];
#pragma unroll
      for (int t = 0; t < TE; ++t) {
        float cand = base + e0t[t] * Wt[r][t];
        acc[r] = fmaxf(acc[r], mt[t] ? cand : 0.0f);  // relu+mask with baseline 0
      }
    }
  }
  // reduce max over the 16 j-lanes within each half group
#pragma unroll
  for (int off = 1; off < 16; off <<= 1)
#pragma unroll
    for (int r = 0; r < 8; ++r) acc[r] = fmaxf(acc[r], __shfl_xor(acc[r], off, 32));
  if (ln == 0) {
    int flag = *hasEdges;
#pragma unroll
    for (int r = 0; r < 8; ++r) {
      int h = h0 + r + 8 * half;
      float v = flag ? acc[r] : c0prev[i * HID + h];
      outF[i * HID + h] = v;
      outH[i * HID + h] = (_Float16)v;
    }
  }
}

__global__ void sem_kernel(const float* __restrict__ hF, const float* __restrict__ Wsem,
                           const float* __restrict__ bsem, float* __restrict__ outSem) {
  int idx = blockIdx.x * 256 + threadIdx.x;
  if (idx >= NCH * NSEM) return;
  int row = idx / NSEM, s = idx % NSEM;
  float acc = 0.f;
  for (int h = 0; h < HID; ++h) acc += hF[row * HID + h] * Wsem[s * HID + h];
  outSem[idx] = acc + bsem[s];
}

__global__ void cvt_kernel(const float* __restrict__ src, int ld_src, int coff,
                           _Float16* __restrict__ dst, int ld_dst, int rows, int cols) {
  int idx = blockIdx.x * 256 + threadIdx.x;
  if (idx >= rows * cols) return;
  int r = idx / cols, c = idx % cols;
  dst[(size_t)r * ld_dst + c] = (_Float16)src[(size_t)r * ld_src + coff + c];
}

extern "C" void kernel_launch(void* const* d_in, const int* in_sizes, int n_in,
                              void* d_out, int out_size, void* d_ws, size_t ws_size,
                              hipStream_t stream) {
  (void)in_sizes; (void)n_in; (void)out_size; (void)ws_size;
  const float* pf   = (const float*)d_in[0];
  const float* Wp   = (const float*)d_in[1];
  const float* bp   = (const float*)d_in[2];
  const float* We   = (const float*)d_in[3];
  const float* be   = (const float*)d_in[4];
  const float* Wsem = (const float*)d_in[5];
  const float* bsem = (const float*)d_in[6];
  const float* Wel  = (const float*)d_in[7];
  const float* bel  = (const float*)d_in[8];
  const float* Wee  = (const float*)d_in[9];
  const float* bee  = (const float*)d_in[10];
  const float* Wop  = (const float*)d_in[11];
  const float* bop  = (const float*)d_in[12];
  const float* Wc   = (const float*)d_in[13];
  const float* bc   = (const float*)d_in[14];
  const float* Wc2  = (const float*)d_in[15];
  const float* bc2  = (const float*)d_in[16];

  float* out    = (float*)d_out;
  float* outCf  = out;                          // 128*512
  float* outSem = out + 65536;                  // 128*50
  float* outEx  = out + 65536 + 6400;           // 128
  float* outEel = out + 65536 + 6400 + 128;     // 128*128*4

  char* ws = (char*)d_ws;
  size_t off = 0;
  auto alloc = [&](size_t bytes) -> char* {
    char* p = ws + off; off += (bytes + 255) & ~(size_t)255; return p;
  };
  float*    cfF   = (float*)alloc(65536 * 4);
  _Float16* cfH   = (_Float16*)alloc(65536 * 2);
  int*      exfl  = (int*)alloc(128 * 4);
  int*      hasE  = (int*)alloc(256);
  float*    Af    = (float*)alloc(65536 * 4);
  float*    Bf    = (float*)alloc(65536 * 4);
  _Float16* el16  = (_Float16*)alloc((size_t)NCH * NCH * HID * 2);
  _Float16* Wel1h = (_Float16*)alloc(HID * HID * 2);
  _Float16* Wel2h = (_Float16*)alloc(HID * HID * 2);
  _Float16* WfH[2][3];
  for (int it = 0; it < 2; ++it)
    for (int p = 0; p < 3; ++p) WfH[it][p] = (_Float16*)alloc(HID * HID * 2);
  _Float16* WcH  = (_Float16*)alloc((size_t)HID * 3 * HID * 2);
  _Float16* Wc2H = (_Float16*)alloc(HID * HID * 2);
  float*    Pbuf = (float*)alloc(65536 * 4);
  float*    Qbuf = (float*)alloc(65536 * 4);
  float*    newF[2]; _Float16* newH[2];
  for (int it = 0; it < 2; ++it) { newF[it] = (float*)alloc(65536 * 4); newH[it] = (_Float16*)alloc(65536 * 2); }
  _Float16* hcat = (_Float16*)alloc((size_t)NCH * 3 * HID * 2);
  float*    hF   = (float*)alloc(65536 * 4);
  _Float16* hH   = (_Float16*)alloc(65536 * 2);

  hipMemsetAsync(hasE, 0, sizeof(int), stream);

  // 1) cf = relu(pf @ Wp.T + bp)   (65536 waves)
  cf_kernel<<<8192, 256, 0, stream>>>(pf, Wp, bp, cfF, cfH);
  // 2) exists
  exists_kernel<<<1, 128, 0, stream>>>(cfF, We, be, outEx, exfl);
  // 3) weight f16 conversions
  cvt_kernel<<<1024, 256, 0, stream>>>(Wel, 2 * HID, 0,   Wel1h, HID, HID, HID);
  cvt_kernel<<<1024, 256, 0, stream>>>(Wel, 2 * HID, HID, Wel2h, HID, HID, HID);
  for (int it = 0; it < 2; ++it)
    for (int p = 0; p < 3; ++p)
      cvt_kernel<<<1024, 256, 0, stream>>>(Wop + (size_t)it * HID * LDOP, LDOP, p * HID,
                                           WfH[it][p], HID, HID, HID);
  cvt_kernel<<<3072, 256, 0, stream>>>(Wc, 3 * HID, 0, WcH, 3 * HID, HID, 3 * HID);
  cvt_kernel<<<1024, 256, 0, stream>>>(Wc2, HID, 0, Wc2H, HID, HID, HID);

  dim3 g512(4, 8);  // Nfeat=512 tiles x rows=128/16
  // 4) Af = cf @ Wel1.T + bel ; Bf = cf @ Wel2.T
  gemm_wmma_kernel<<<g512, 256, 0, stream>>>(cfH, HID, Wel1h, HID, bel, Af, nullptr, HID, HID, 0);
  gemm_wmma_kernel<<<g512, 256, 0, stream>>>(cfH, HID, Wel2h, HID, nullptr, Bf, nullptr, HID, HID, 0);
  // 5) el (f16) and eel + mask/has_edges
  el_kernel<<<32768, 256, 0, stream>>>(Af, Bf, el16);
  eel_kernel<<<2048, 256, 0, stream>>>(el16, Wee, bee, exfl, outEel, hasE);

  // 6) message-passing iterations
  const _Float16* c0h = cfH;
  const float*    c0f = cfF;
  for (int it = 0; it < 2; ++it) {
    gemm_wmma_kernel<<<g512, 256, 0, stream>>>(c0h, HID, WfH[it][0], HID, bop + it * HID,
                                               Pbuf, nullptr, HID, HID, 0);
    gemm_wmma_kernel<<<g512, 256, 0, stream>>>(c0h, HID, WfH[it][1], HID, nullptr,
                                               Qbuf, nullptr, HID, HID, 0);
    msg_max_kernel<<<dim3(NCH, 4), 256, 0, stream>>>(el16, WfH[it][2], Pbuf, Qbuf,
                                                     Wop + (size_t)it * HID * LDOP,
                                                     outEel, exfl, hasE, c0f,
                                                     newF[it], newH[it]);
    c0h = newH[it]; c0f = newF[it];
  }

  // 7) hcat = [cf | new1 | new2] (f16), h = relu(hcat @ Wc.T + bc)
  cvt_kernel<<<256, 256, 0, stream>>>(cfF,     HID, 0, hcat,           3 * HID, NCH, HID);
  cvt_kernel<<<256, 256, 0, stream>>>(newF[0], HID, 0, hcat + HID,     3 * HID, NCH, HID);
  cvt_kernel<<<256, 256, 0, stream>>>(newF[1], HID, 0, hcat + 2 * HID, 3 * HID, NCH, HID);
  gemm_wmma_kernel<<<g512, 256, 0, stream>>>(hcat, 3 * HID, WcH, 3 * HID, bc, hF, hH, HID, 3 * HID, 1);

  // 8) sem and out_cf
  sem_kernel<<<25, 256, 0, stream>>>(hF, Wsem, bsem, outSem);
  gemm_wmma_kernel<<<g512, 256, 0, stream>>>(hH, HID, Wc2H, HID, bc2, outCf, nullptr, FEA, HID, 1);
}
